// DeepGANDiscriminator_70798240907332
// MI455X (gfx1250) — matrix operand order repaired and checked by reading, not compiled, not run on hardware
//
#include <hip/hip_runtime.h>
#include <stdint.h>
#include <stddef.h>

// ---------------- types ----------------
typedef __attribute__((ext_vector_type(16))) __bf16 v16bf;
typedef __attribute__((ext_vector_type(8)))  float  v8f;
typedef __attribute__((ext_vector_type(4)))  unsigned int u32x4;
typedef __attribute__((ext_vector_type(8)))  int    i32x8;
typedef __attribute__((ext_vector_type(4)))  int    i32x4;

#define BB 64
#define TT 1024

union Frag32B { u32x4 q[2]; v16bf v; };

// A-fragment (16x32 bf16, M x K): lane<16 holds row M=lane&15, K = k0..k0+7 and
// k0+16..k0+23 ; lane>=16 holds same row, K shifted by +8. (ISA 7.12.2)
static __device__ inline v16bf load_afrag(const __bf16* X, int ldk, int row, int k0, int lane) {
  const int hi = (lane >= 16) ? 8 : 0;
  const __bf16* p = X + (size_t)row * ldk + (k0 + hi);
  Frag32B f;
  f.q[0] = *(const u32x4*)(p);
  f.q[1] = *(const u32x4*)(p + 16);
  return f.v;
}

// B-fragment (32x16 bf16, K x N) built from row-major W[col][k]:
// lane n holds column N0+(n&15); lanes<16 carry K=k0..k0+7,k0+16..k0+23, lanes>=16 K shifted by +8.
static __device__ inline v16bf load_bfrag(const __bf16* W, int ldk, int col, int k0, int lane) {
  const int hi = (lane >= 16) ? 8 : 0;
  const __bf16* p = W + (size_t)col * ldk + (k0 + hi);
  Frag32B f;
  f.q[0] = *(const u32x4*)(p);
  f.q[1] = *(const u32x4*)(p + 16);
  return f.v;
}

// B-fragment from LDS, pre-swizzled: frag f, lane ln -> 32 contiguous bytes.
static __device__ inline v16bf load_bfrag_lds(const __bf16* wlds, int f, int lane) {
  const u32x4* p = (const u32x4*)(wlds + ((size_t)f * 32 + lane) * 16);
  Frag32B fr;
  fr.q[0] = p[0];
  fr.q[1] = p[1];
  return fr.v;
}

static __device__ inline float sigmoidf_fast(float x) {
  return 1.0f / (1.0f + __expf(-x));
}

// ---------------- prep kernels ----------------
__global__ void pack_x_kernel(const float* __restrict__ x, const float* __restrict__ lab,
                              __bf16* __restrict__ xcat) {
  // xcat: [B*T][160] bf16 ; x: [B*T][128] ; lab: [B*T][32]
  size_t total = (size_t)BB * TT * 160;
  for (size_t i = (size_t)blockIdx.x * blockDim.x + threadIdx.x; i < total;
       i += (size_t)gridDim.x * blockDim.x) {
    size_t row = i / 160;
    int c = (int)(i % 160);
    float v = (c < 128) ? x[row * 128 + c] : lab[row * 32 + (c - 128)];
    xcat[i] = (__bf16)v;
  }
}

__global__ void cvt_bf16_kernel(const float* __restrict__ src, __bf16* __restrict__ dst, size_t n) {
  for (size_t i = (size_t)blockIdx.x * blockDim.x + threadIdx.x; i < n;
       i += (size_t)gridDim.x * blockDim.x) {
    dst[i] = (__bf16)src[i];
  }
}

// ---------------- input-projection GEMM: GI = X * W^T + b ----------------
// X: [B*T][KIN] bf16 row-major; W: [NCOLS][KIN] bf16 row-major; GI: [B*T][NCOLS] f32
// Block = 256 threads (8 waves). Wave w computes rows [blockIdx.x*256 + w*32, +32),
// cols [blockIdx.y*64, +64): 2 A-frags x 4 B-frags -> 8 WMMA accumulators, so each
// B-fragment load feeds two WMMAs.
template<int KIN, int NCOLS>
__global__ void __launch_bounds__(256) gi_gemm_kernel(const __bf16* __restrict__ X,
                                                      const __bf16* __restrict__ W,
                                                      const float*  __restrict__ bias,
                                                      float* __restrict__ GI) {
  const int lane = threadIdx.x & 31;
  const int wave = threadIdx.x >> 5;
  const int M0 = blockIdx.x * 256 + wave * 32;
  const int N0 = blockIdx.y * 64;

  v8f acc0[4] = {v8f{}, v8f{}, v8f{}, v8f{}};
  v8f acc1[4] = {v8f{}, v8f{}, v8f{}, v8f{}};
  for (int k0 = 0; k0 < KIN; k0 += 32) {
    v16bf a0 = load_afrag(X, KIN, M0 + (lane & 15), k0, lane);
    v16bf a1 = load_afrag(X, KIN, M0 + 16 + (lane & 15), k0, lane);
#pragma unroll
    for (int c = 0; c < 4; ++c) {
      v16bf b = load_bfrag(W, KIN, N0 + c * 16 + (lane & 15), k0, lane);
      acc0[c] = __builtin_amdgcn_wmma_f32_16x16x32_bf16(false, a0, false, b,
                                                        (short)0, acc0[c], false, false);
      acc1[c] = __builtin_amdgcn_wmma_f32_16x16x32_bf16(false, a1, false, b,
                                                        (short)0, acc1[c], false, false);
    }
  }
  // C/D layout: lane<16 -> N=lane, M=M0+v (v=0..7); lane>=16 -> N=lane-16, M=M0+8+v
#pragma unroll
  for (int c = 0; c < 4; ++c) {
    const int col = N0 + c * 16 + (lane & 15);
    const float bv = bias[col];
#pragma unroll
    for (int v = 0; v < 8; ++v) {
      const int r0 = M0 + ((lane < 16) ? v : v + 8);
      GI[(size_t)r0 * NCOLS + col] = acc0[c][v] + bv;
      GI[(size_t)(r0 + 16) * NCOLS + col] = acc1[c][v] + bv;
    }
  }
}

// ---------------- persistent GRU recurrence ----------------
// GI:  [B*T][3H] f32 (precomputed input-side gates)
// Whh: [3H][H] bf16 row-major ; bhh: [3H] f32
// Hout:[B][T][H] bf16 (layer output = h_t for every t)
// Grid = H/64 workgroups; WG g owns hidden units [g*64, g*64+64).
// LDS: hbuf = full h state (64 x H bf16, refreshed per step via TDM tile load) +
//      wlds = this WG's W_hh slice (192 x H bf16) staged ONCE, pre-swizzled into
//      exact B-fragment order so the steady-state loop is ds_load_b128 -> v_wmma.
template<int H>
__global__ void __launch_bounds__(256) gru_rec_kernel(const float*  __restrict__ GI,
                                                      const __bf16* __restrict__ Whh,
                                                      const float*  __restrict__ bhh,
                                                      __bf16* __restrict__ Hout,
                                                      unsigned int* __restrict__ syncv) {
  constexpr int NWG = H / 64;
  constexpr int KC = H / 32;          // number of 32-wide K chunks
  extern __shared__ __bf16 hbuf[];    // [64][H] bf16 at LDS offset 0
  __bf16* wlds = hbuf + 64 * H;       // [12 col-tiles][KC][32 lanes][16 bf16]
  const int lane = threadIdx.x & 31;
  const int wave = threadIdx.x >> 5;
  const int wgu0 = blockIdx.x * 64;

  // ---- one-time staging: zero h0, swizzle W_hh slice into LDS ----
  for (int i = threadIdx.x; i < (64 * H) / 8; i += 256) {
    ((u32x4*)hbuf)[i] = u32x4{0u, 0u, 0u, 0u};
  }
  for (int fl = threadIdx.x; fl < 12 * KC * 32; fl += 256) {
    const int ln = fl & 31;
    const int f  = fl >> 5;          // fragment index: ct*KC + kc
    const int kc = f % KC;
    const int ct = f / KC;           // 0..11 ; gate = ct>>2, unit-tile = ct&3
    const int g  = ct >> 2;
    const int ut = ct & 3;
    const int col = g * H + wgu0 + ut * 16 + (ln & 15);
    const int k0  = kc * 32 + ((ln >= 16) ? 8 : 0);
    const u32x4* src = (const u32x4*)(Whh + (size_t)col * H + k0);
    u32x4* dst = (u32x4*)(wlds + ((size_t)f * 32 + ln) * 16);
    dst[0] = src[0];
    dst[1] = src[2];
  }
  __syncthreads();

  for (int t = 0; t < TT; ++t) {
    // ---- stage h_{t-1} into LDS via TDM (2D tile: 64 rows x H bf16, stride T*H) ----
    if (t > 0) {
      if (wave == 0) {
        const uint64_t gaddr = (uint64_t)(uintptr_t)(Hout + (size_t)(t - 1) * H);
        u32x4 g0;
        g0[0] = 1u;  // count=1, user mode, no gather
        g0[1] = 0u;  // lds_addr = 0 (hbuf at base of dynamic LDS)
        g0[2] = (unsigned)(gaddr & 0xFFFFFFFFu);
        g0[3] = (unsigned)((gaddr >> 32) & 0x1FFFFFFu) | (2u << 30);  // type=2
        i32x8 g1;
        g1[0] = (int)(1u << 16);                          // data_size=2B; no multicast
        g1[1] = (int)(((unsigned)H & 0xFFFFu) << 16);     // tensor_dim0 lo16 = H
        g1[2] = (int)(((unsigned)H >> 16) | (64u << 16)); // dim0 hi | tensor_dim1 lo16=64
        g1[3] = (int)((unsigned)H << 16);                 // dim1 hi=0 | tile_dim0 = H
        g1[4] = 64;                                       // tile_dim1=64, tile_dim2=0
        const unsigned long long s0 = (unsigned long long)TT * (unsigned long long)H;
        g1[5] = (int)(s0 & 0xFFFFFFFFu);                  // tensor_dim0_stride lo32
        g1[6] = (int)((s0 >> 32) & 0xFFFFu);              // stride hi16 | dim1_stride=0
        g1[7] = 0;
        i32x4 gz4 = {0, 0, 0, 0};
        i32x8 gz8 = {0, 0, 0, 0, 0, 0, 0, 0};
        __builtin_amdgcn_tensor_load_to_lds(g0, g1, gz4, gz4, gz8, 0);
        __builtin_amdgcn_s_wait_tensorcnt(0);
      }
      __syncthreads();
    }

    // ---- gate GEMM + pointwise: 2 (row-tile, unit-tile) combos per wave ----
#pragma unroll
    for (int combo = 0; combo < 2; ++combo) {
      const int idx = wave * 2 + combo;  // 0..15
      const int rt = idx & 3;            // batch-row tile (16 rows)
      const int ut = idx >> 2;           // hidden-unit tile (16 units)
      const int M0 = rt * 16;
      const int u0 = ut * 16;

      v8f ar = v8f{}, az = v8f{}, an = v8f{};
      for (int kc = 0; kc < KC; ++kc) {
        const int k0 = kc * 32;
        v16bf a = load_afrag(hbuf, H, M0 + (lane & 15), k0, lane);
        v16bf br = load_bfrag_lds(wlds, (0 * 4 + ut) * KC + kc, lane);
        ar = __builtin_amdgcn_wmma_f32_16x16x32_bf16(false, a, false, br, (short)0, ar, false, false);
        v16bf bz = load_bfrag_lds(wlds, (1 * 4 + ut) * KC + kc, lane);
        az = __builtin_amdgcn_wmma_f32_16x16x32_bf16(false, a, false, bz, (short)0, az, false, false);
        v16bf bn = load_bfrag_lds(wlds, (2 * 4 + ut) * KC + kc, lane);
        an = __builtin_amdgcn_wmma_f32_16x16x32_bf16(false, a, false, bn, (short)0, an, false, false);
      }

      const int unit = wgu0 + u0 + (lane & 15);
      const float brv = bhh[0 * H + unit];
      const float bzv = bhh[1 * H + unit];
      const float bnv = bhh[2 * H + unit];
#pragma unroll
      for (int v = 0; v < 8; ++v) {
        const int brow = M0 + ((lane < 16) ? v : v + 8);  // batch index
        const size_t gib = ((size_t)brow * TT + t) * (size_t)(3 * H);
        const float ir  = GI[gib + 0 * H + unit];
        const float iz  = GI[gib + 1 * H + unit];
        const float inn = GI[gib + 2 * H + unit];
        const float r = sigmoidf_fast(ir + ar[v] + brv);
        const float z = sigmoidf_fast(iz + az[v] + bzv);
        const float n = tanhf(inn + r * (an[v] + bnv));
        const float hprev = (float)hbuf[(size_t)brow * H + unit];
        const float hnew = (1.0f - z) * n + z * hprev;
        Hout[((size_t)brow * TT + t) * H + unit] = (__bf16)hnew;
      }
    }

    // ---- grid sync (monotonic counter; NWG <= 8 so all WGs are resident) ----
    __threadfence();
    __syncthreads();
    if (threadIdx.x == 0) {
      __hip_atomic_fetch_add(syncv, 1u, __ATOMIC_RELEASE, __HIP_MEMORY_SCOPE_AGENT);
      const unsigned target = (unsigned)NWG * (unsigned)(t + 1);
      while (__hip_atomic_load(syncv, __ATOMIC_ACQUIRE, __HIP_MEMORY_SCOPE_AGENT) < target) {
        __builtin_amdgcn_s_sleep(1);
      }
    }
    __syncthreads();
  }
}

// ---------------- final FC: out[b] = dot(h3[b, T-1, :], fc_w) + fc_b ----------------
__global__ void fc_kernel(const __bf16* __restrict__ h3, const float* __restrict__ fcw,
                          const float* __restrict__ fcb, float* __restrict__ out) {
  const int b = threadIdx.x;
  if (b < BB) {
    const __bf16* hp = h3 + ((size_t)b * TT + (TT - 1)) * 128;
    float s = 0.0f;
#pragma unroll 8
    for (int k = 0; k < 128; ++k) s += (float)hp[k] * fcw[k];
    out[b] = s + fcb[0];
  }
}

// ---------------- launch ----------------
extern "C" void kernel_launch(void* const* d_in, const int* in_sizes, int n_in,
                              void* d_out, int out_size, void* d_ws, size_t ws_size,
                              hipStream_t stream) {
  (void)in_sizes; (void)n_in; (void)out_size; (void)ws_size;

  const float* x    = (const float*)d_in[0];
  const float* lab  = (const float*)d_in[1];
  const float* wih1 = (const float*)d_in[2];
  const float* whh1 = (const float*)d_in[3];
  const float* bih1 = (const float*)d_in[4];
  const float* bhh1 = (const float*)d_in[5];
  const float* wih2 = (const float*)d_in[6];
  const float* whh2 = (const float*)d_in[7];
  const float* bih2 = (const float*)d_in[8];
  const float* bhh2 = (const float*)d_in[9];
  const float* wih3 = (const float*)d_in[10];
  const float* whh3 = (const float*)d_in[11];
  const float* bih3 = (const float*)d_in[12];
  const float* bhh3 = (const float*)d_in[13];
  const float* fcw  = (const float*)d_in[14];
  const float* fcb  = (const float*)d_in[15];

  char* cur = (char*)d_ws;
  auto take = [&](size_t bytes) -> char* {
    char* p = cur;
    cur += (bytes + 255) & ~(size_t)255;
    return p;
  };

  __bf16* xcat  = (__bf16*)take((size_t)BB * TT * 160 * 2);
  __bf16* wih1b = (__bf16*)take((size_t)1536 * 160 * 2);
  __bf16* whh1b = (__bf16*)take((size_t)1536 * 512 * 2);
  __bf16* wih2b = (__bf16*)take((size_t)768 * 512 * 2);
  __bf16* whh2b = (__bf16*)take((size_t)768 * 256 * 2);
  __bf16* wih3b = (__bf16*)take((size_t)384 * 256 * 2);
  __bf16* whh3b = (__bf16*)take((size_t)384 * 128 * 2);
  float*  gi    = (float*)take((size_t)BB * TT * 1536 * 4);  // reused by all 3 layers
  __bf16* h1    = (__bf16*)take((size_t)BB * TT * 512 * 2);
  __bf16* h2    = (__bf16*)take((size_t)BB * TT * 256 * 2);
  __bf16* h3    = (__bf16*)take((size_t)BB * TT * 128 * 2);
  unsigned int* syncv = (unsigned int*)take(256);

  (void)hipMemsetAsync(syncv, 0, 256, stream);

  pack_x_kernel<<<512, 256, 0, stream>>>(x, lab, xcat);
  cvt_bf16_kernel<<<120, 256, 0, stream>>>(wih1, wih1b, (size_t)1536 * 160);
  cvt_bf16_kernel<<<120, 256, 0, stream>>>(whh1, whh1b, (size_t)1536 * 512);
  cvt_bf16_kernel<<<120, 256, 0, stream>>>(wih2, wih2b, (size_t)768 * 512);
  cvt_bf16_kernel<<<120, 256, 0, stream>>>(whh2, whh2b, (size_t)768 * 256);
  cvt_bf16_kernel<<<120, 256, 0, stream>>>(wih3, wih3b, (size_t)384 * 256);
  cvt_bf16_kernel<<<120, 256, 0, stream>>>(whh3, whh3b, (size_t)384 * 128);

  // Layer 1: In=160, H=512   (LDS: 64KB h + 192KB W = 256KB < 320KB WGP LDS)
  gi_gemm_kernel<160, 1536><<<dim3(256, 24), 256, 0, stream>>>(xcat, wih1b, bih1, gi);
  gru_rec_kernel<512><<<8, 256, (64 + 192) * 512 * 2, stream>>>(gi, whh1b, bhh1, h1, syncv + 0);
  // Layer 2: In=512, H=256   (LDS: 128KB)
  gi_gemm_kernel<512, 768><<<dim3(256, 12), 256, 0, stream>>>(h1, wih2b, bih2, gi);
  gru_rec_kernel<256><<<4, 256, (64 + 192) * 256 * 2, stream>>>(gi, whh2b, bhh2, h2, syncv + 16);
  // Layer 3: In=256, H=128   (LDS: 64KB)
  gi_gemm_kernel<256, 384><<<dim3(256, 6), 256, 0, stream>>>(h2, wih3b, bih3, gi);
  gru_rec_kernel<128><<<2, 256, (64 + 192) * 128 * 2, stream>>>(gi, whh3b, bhh3, h3, syncv + 32);

  fc_kernel<<<1, 64, 0, stream>>>(h3, fcw, fcb, (float*)d_out);
}
